// HashSoftmax_57629871178518
// MI455X (gfx1250) — matrix operand order repaired and checked by reading, not compiled, not run on hardware
//
#include <hip/hip_runtime.h>

// HashSoftmax for MI455X (gfx1250, wave32, WMMA).
// out[M,V] = x[M,K] @ embed[V,K]^T,  embed[v,:] = sum_j ip[v,j] * pool[hash[v,j],:]
// M = 2*2048 = 4096, K = HIDDEN = 256, V = 32000, POOL = 100000, NHASH = 20.
//
// bf16 WMMA GEMM with f32 accumulation (v_wmma_f32_16x16x32_bf16).
// Roofline: output stream 512MB f32 -> ~22us at 23.3 TB/s; bf16 matrix ops
// bring compute under/near that, so the kernel runs at the memory roof.
// B tile staged into LDS with GLOBAL_LOAD_ASYNC_TO_LDS_B128 (ASYNCcnt path),
// output stored non-temporally to keep pool/embed/x resident in the 192MB L2.

#define HID   256
#define NV    32000
#define NHASH 20

typedef unsigned short ushort_t;
typedef __attribute__((ext_vector_type(16))) __bf16 v16bf;
typedef __attribute__((ext_vector_type(8)))  float  v8f;

#define AS3 __attribute__((address_space(3)))

union Frag16 {
    v16bf v;
    uint4 q[2];
};

static __device__ __forceinline__ ushort_t f32_to_bf16_rne(float f) {
    unsigned int u = __float_as_uint(f);
    unsigned int r = (u + 0x7FFFu + ((u >> 16) & 1u)) >> 16;  // RNE to top 16b
    return (ushort_t)r;
}

// ---------------------------------------------------------------------------
// Kernel 1: convert x (f32) -> bf16, 4 elements per thread, vectorized.
// ---------------------------------------------------------------------------
__global__ __launch_bounds__(256)
void convert_x_bf16(const float* __restrict__ x, ushort_t* __restrict__ xbf, int n) {
    int i = (blockIdx.x * 256 + threadIdx.x) * 4;
    if (i >= n) return;
    float4 f = *(const float4*)(x + i);
    ushort4 o;
    o.x = f32_to_bf16_rne(f.x);
    o.y = f32_to_bf16_rne(f.y);
    o.z = f32_to_bf16_rne(f.z);
    o.w = f32_to_bf16_rne(f.w);
    *(ushort4*)(xbf + i) = o;
}

// ---------------------------------------------------------------------------
// Kernel 2: embed[v][h] = sum_j ip[v][j] * pool[hash[v][j]][h], stored bf16.
// One block per vocab row, 256 threads == HIDDEN. Pool (100MB) is L2-resident
// (192MB L2); gathers are coalesced full-row reads with prefetch of the next
// hashed row (global_prefetch_b8).
// ---------------------------------------------------------------------------
__global__ __launch_bounds__(256)
void build_embed_bf16(const float* __restrict__ pool,
                      const float* __restrict__ ip,
                      const int*   __restrict__ hashv,
                      ushort_t*    __restrict__ emb) {
    __shared__ int   sh[NHASH];
    __shared__ float sw[NHASH];
    const int v   = blockIdx.x;
    const int tid = threadIdx.x;

    if (tid < NHASH) {
        sh[tid] = hashv[v * NHASH + tid];
        sw[tid] = ip[v * NHASH + tid];
    }
    __syncthreads();

    float acc = 0.0f;
#pragma unroll 4
    for (int j = 0; j < NHASH; ++j) {
        const float* row = pool + (size_t)sh[j] * HID;
        if (j + 1 < NHASH) {
            __builtin_prefetch(pool + (size_t)sh[j + 1] * HID + tid, 0, 3);
        }
        acc += sw[j] * row[tid];
    }
    emb[(size_t)v * HID + tid] = f32_to_bf16_rne(acc);
}

// ---------------------------------------------------------------------------
// Kernel 3: GEMM  out[M][NV] = xbf[M][K] * emb[NV][K]^T  (bf16 in, f32 out)
// Block: 256 threads = 8 waves. Block tile: 256(M) x 64(N).
// Wave w owns rows [m0 + 32w, +32) (two 16-row subtiles) and the full 64-wide
// N strip (4 subtiles) -> 8 f32 accumulators, 8 WMMAs per K-step, each B
// fragment feeds 2 WMMAs. B tile (64 x 256 K bf16 = 32KB + pad) staged in LDS
// via async direct-to-LDS loads, shared by all 8 waves. A fragments come
// straight from global (x_bf16 is 2MB -> L2 resident).
// ---------------------------------------------------------------------------
#define BPAD 16   // pad row to 272 halves to spread LDS banks on frag reads

__global__ __launch_bounds__(256)
void gemm_bf16_wmma(const ushort_t* __restrict__ xbf,
                    const ushort_t* __restrict__ emb,
                    float* __restrict__ out) {
    __shared__ ushort_t Bs[64][HID + BPAD];

    const int tid  = threadIdx.x;
    const int wave = tid >> 5;
    const int lane = tid & 31;
    const int n0   = blockIdx.x * 64;               // N tile origin
    const int m0   = blockIdx.y * 256 + wave * 32;  // this wave's M origin

    // ---- async direct-to-LDS stage of the B tile: 64 x 256 halves --------
    // 2048 chunks of 16B; 8 chunks per thread. GVS addressing:
    //   mem = s[base] + voff ; LDS dest = 32-bit LDS byte address.
    {
        const ushort_t* base = emb + (size_t)n0 * HID;
        const unsigned bs_base =
            (unsigned)(__UINTPTR_TYPE__)(AS3 void*)&Bs[0][0];
#pragma unroll
        for (int c = tid; c < 64 * (HID / 8); c += 256) {
            const int row = c >> 5;            // 32 chunks of 8 halves per row
            const int col = (c & 31) * 8;
            const unsigned voff    = (unsigned)(row * HID + col) * 2u;
            const unsigned lds_dst = bs_base + (unsigned)(row * (HID + BPAD) + col) * 2u;
            asm volatile("global_load_async_to_lds_b128 %0, %1, %2"
                         :
                         : "v"(lds_dst), "v"(voff), "s"(base)
                         : "memory");
        }
        asm volatile("s_wait_asynccnt 0x0" ::: "memory");
    }
    __syncthreads();

    v8f acc[8] = {};

    const int hi = lane >> 4;     // 0: lanes 0-15, 1: lanes 16-31
    const int lm = lane & 15;
    const ushort_t* arow0 = xbf + (size_t)(m0 + lm) * HID;        // M subtile 0
    const ushort_t* arow1 = xbf + (size_t)(m0 + 16 + lm) * HID;   // M subtile 1

    // K = 256 -> 8 steps of 32
#pragma unroll
    for (int kk = 0; kk < 8; ++kk) {
        const int k0 = kk * 32;

        // A fragments (ISA 7.12.2, 16-bit A 16x32):
        //   lanes 0-15 : elems 0-7 = K k0+0..7,  8-15 = K k0+16..23
        //   lanes 16-31: elems 0-7 = K k0+8..15, 8-15 = K k0+24..31
        Frag16 a0, a1;
        a0.q[0] = *(const uint4*)(arow0 + k0 + hi * 8);
        a0.q[1] = *(const uint4*)(arow0 + k0 + 16 + hi * 8);
        a1.q[0] = *(const uint4*)(arow1 + k0 + hi * 8);
        a1.q[1] = *(const uint4*)(arow1 + k0 + 16 + hi * 8);

#pragma unroll
        for (int nt = 0; nt < 4; ++nt) {
            // B fragment (32x16, 16-bit): same K order as A; lanes 0-15 ->
            // K k0+0..7, k0+16..23; lanes 16-31 -> K k0+8..15, k0+24..31
            Frag16 b;
            const ushort_t* bp = &Bs[nt * 16 + lm][k0 + hi * 8];
            b.q[0] = *(const uint4*)(bp);
            b.q[1] = *(const uint4*)(bp + 16);

            acc[nt] = __builtin_amdgcn_wmma_f32_16x16x32_bf16(
                false, a0.v, false, b.v, (short)0, acc[nt], false, false);
            acc[4 + nt] = __builtin_amdgcn_wmma_f32_16x16x32_bf16(
                false, a1.v, false, b.v, (short)0, acc[4 + nt], false, false);
        }
    }

    // ---- store C/D: VGPR r -> (M = msub + r + 8*hi, N = n0 + nt*16 + lm) --
    // Non-temporal: 512MB pure stream; keep L2 for pool/embed/x.
#pragma unroll
    for (int nt = 0; nt < 4; ++nt) {
        const int col = n0 + nt * 16 + lm;
#pragma unroll
        for (int r = 0; r < 8; ++r) {
            const int row0 = m0 + r + 8 * hi;
            __builtin_nontemporal_store(acc[nt][r],
                                        out + (size_t)row0 * NV + col);
            const int row1 = m0 + 16 + r + 8 * hi;
            __builtin_nontemporal_store(acc[4 + nt][r],
                                        out + (size_t)row1 * NV + col);
        }
    }
}

// ---------------------------------------------------------------------------
// Host-side launch. Inputs (setup_inputs order):
//   d_in[0] = x             f32 [2,2048,256]
//   d_in[1] = pool          f32 [100000,256]
//   d_in[2] = import_params f32 [32000,20]
//   d_in[3] = hash_values   int [32000,20]
// d_out = f32 [2,2048,32000]
// Workspace: xbf (M*256*2 = 2MB) + emb (32000*256*2 = 16MB) = 18MB.
// ---------------------------------------------------------------------------
extern "C" void kernel_launch(void* const* d_in, const int* in_sizes, int n_in,
                              void* d_out, int out_size, void* d_ws, size_t ws_size,
                              hipStream_t stream) {
    const float* x     = (const float*)d_in[0];
    const float* pool  = (const float*)d_in[1];
    const float* ip    = (const float*)d_in[2];
    const int*   hashv = (const int*)d_in[3];
    float*       out   = (float*)d_out;

    const int M = in_sizes[0] / HID;           // 4096
    ushort_t* xbf = (ushort_t*)d_ws;
    ushort_t* emb = xbf + (size_t)M * HID;

    // 1) x -> bf16
    {
        int n = M * HID;
        convert_x_bf16<<<(n / 4 + 255) / 256, 256, 0, stream>>>(x, xbf, n);
    }
    // 2) hashed-embedding build -> bf16
    build_embed_bf16<<<NV, HID, 0, stream>>>(pool, ip, hashv, emb);
    // 3) logits GEMM (bf16 WMMA, f32 accumulate)
    {
        dim3 grid(NV / 64, M / 256);
        gemm_bf16_wmma<<<grid, 256, 0, stream>>>(xbf, emb, out);
    }
}
